// Attention_all_56736517980392
// MI455X (gfx1250) — matrix operand order repaired and checked by reading, not compiled, not run on hardware
//
#include <hip/hip_runtime.h>
#include <hip/hip_bf16.h>

// ---------------------------------------------------------------------------
// Softmax-free attention block for MI455X (gfx1250, wave32, WMMA).
//   y = x + out + mlp,  out = q(k^T v)   [associativity: no softmax => exact]
// GEMMs on v_wmma_f32_16x16x32_f16 (f16 in, f32 accumulate).
// Big GEMMs stage A/B tiles in LDS with GLOBAL_LOAD_ASYNC_TO_LDS_B128,
// double-buffered, synchronized via s_wait_asynccnt + workgroup barrier.
// ---------------------------------------------------------------------------

typedef __attribute__((ext_vector_type(16))) _Float16 v16h;
typedef __attribute__((ext_vector_type(8)))  _Float16 v8h;
typedef __attribute__((ext_vector_type(8)))  float    v8f;

static constexpr int BATCH  = 2;
static constexpr int SEQ    = 2048;
static constexpr int TOK    = BATCH * SEQ;   // 4096
static constexpr int DIMC   = 1024;
static constexpr int HEADS  = 16;
static constexpr int HD     = 64;
static constexpr float QSCALE = 0.125f;      // 64^-0.5

static constexpr int TPAD = 40;  // LDS tile row pitch in halves (80B: 20-bank
                                 // stride => conflict-free b128 lane reads)

// ---------------- CDNA5 async copy: global -> LDS (ASYNCcnt) ----------------
__device__ __forceinline__ void async_g2l_b128(const _Float16* g, _Float16* l) {
  asm volatile("global_load_async_to_lds_b128 %0, %1, off"
               :: "v"((unsigned)(unsigned long long)(uintptr_t)l),
                  "v"((unsigned long long)(uintptr_t)g)
               : "memory");
}
__device__ __forceinline__ void wait_async0() {
  asm volatile("s_wait_asynccnt 0x0" ::: "memory");
}

// Stage a 128-row x 32-half tile (rows rowbase..rowbase+127, cols k0..k0+31)
// into padded LDS tile (pitch TPAD). 512 b128 chunks over 256 threads.
__device__ __forceinline__ void stage_tile_async(const _Float16* __restrict__ src,
                                                 int ld, int rowbase, int k0,
                                                 _Float16* dst, int tid) {
#pragma unroll
  for (int i = 0; i < 2; ++i) {
    const int c   = tid + i * 256;     // 0..511
    const int row = c >> 2;            // 0..127
    const int ch  = c & 3;             // 0..3 (8-half chunks)
    async_g2l_b128(src + (size_t)(rowbase + row) * ld + k0 + ch * 8,
                   dst + row * TPAD + ch * 8);
  }
}

// ------------- fragment builders (ISA 7.12.2 16-bit layouts) ---------------
__device__ __forceinline__ v16h combine(v8h lo, v8h hi) {
  v16h r;
#pragma unroll
  for (int i = 0; i < 8; ++i) { r[i] = lo[i]; r[8 + i] = hi[i]; }
  return r;
}

// A (16x32 MxK): lane l -> row mbase+(l&15); K chunks at (l>>4)*8 and 16+(l>>4)*8
__device__ __forceinline__ v16h a_frag_ld(const _Float16* p_row0, int pitch,
                                          int mbase, int lane) {
  const _Float16* p = p_row0 + (size_t)(mbase + (lane & 15)) * pitch
                             + ((lane >> 4) << 3);
  return combine(*(const v8h*)p, *(const v8h*)(p + 16));
}
// B (32x16 KxN): lane l -> col nbase+(l&15); K chunks at (l>>4)*8 and 16+(l>>4)*8
__device__ __forceinline__ v16h b_frag_ld(const _Float16* p_row0, int pitch,
                                          int nbase, int lane) {
  const _Float16* p = p_row0 + (size_t)(nbase + (lane & 15)) * pitch
                             + ((lane >> 4) << 3);
  return combine(*(const v8h*)p, *(const v8h*)(p + 16));
}

__device__ __forceinline__ v8f wmma_f16(v16h a, v16h b, v8f c) {
  return __builtin_amdgcn_wmma_f32_16x16x32_f16(false, a, false, b,
                                                (short)0, c, false, false);
}

__device__ __forceinline__ float gelu_exact(float v) {
  return 0.5f * v * (1.0f + erff(v * 0.70710678118654752440f));
}

// ---------------------------- f32 -> f16 convert ----------------------------
__global__ void cvt_f32_to_f16(const float* __restrict__ s,
                               _Float16* __restrict__ d, int n) {
  int i = blockIdx.x * blockDim.x + threadIdx.x;
  const int stride = gridDim.x * blockDim.x;
  for (; i < n; i += stride) d[i] = (_Float16)s[i];
}

// ------------------- main GEMM: C = (A @ W^T + bias) ops --------------------
// 256 thr (8 waves). Block tile 128x128; wave tile 32x64. A/B slabs staged in
// LDS via async copies, double-buffered across the K loop.
__global__ __launch_bounds__(256) void gemm_wmma_128x128(
    const _Float16* __restrict__ A, int lda,
    const _Float16* __restrict__ W, int ldw,
    const float* __restrict__ bias,
    float scale, int K, int ldo, int do_gelu,
    _Float16* __restrict__ out16,
    float*    __restrict__ out32,
    const float* __restrict__ res0,
    const float* __restrict__ res1,
    _Float16* __restrict__ outT) {
  __shared__ _Float16 As[2][128 * TPAD];   // 10 KB each
  __shared__ _Float16 Bs[2][128 * TPAD];

  const int tid   = threadIdx.x;
  const int lane  = tid & 31;
  const int wave  = tid >> 5;
  const int mtile = blockIdx.y * 128;
  const int ntile = blockIdx.x * 128;
  const int mw    = (wave >> 1) * 32;      // wave row offset in tile
  const int nw    = (wave & 1) * 64;       // wave col offset in tile

  v8f acc[2][4] = {};

  stage_tile_async(A, lda, mtile, 0, As[0], tid);
  stage_tile_async(W, ldw, ntile, 0, Bs[0], tid);
  wait_async0();
  __syncthreads();

  int buf = 0;
  for (int k0 = 0; k0 < K; k0 += 32) {
    const int nxt = buf ^ 1;
    if (k0 + 32 < K) {
      stage_tile_async(A, lda, mtile, k0 + 32, As[nxt], tid);
      stage_tile_async(W, ldw, ntile, k0 + 32, Bs[nxt], tid);
    }
    v16h a0 = a_frag_ld(As[buf], TPAD, mw,      lane);
    v16h a1 = a_frag_ld(As[buf], TPAD, mw + 16, lane);
    v16h b0 = b_frag_ld(Bs[buf], TPAD, nw,      lane);
    v16h b1 = b_frag_ld(Bs[buf], TPAD, nw + 16, lane);
    v16h b2 = b_frag_ld(Bs[buf], TPAD, nw + 32, lane);
    v16h b3 = b_frag_ld(Bs[buf], TPAD, nw + 48, lane);
    acc[0][0] = wmma_f16(a0, b0, acc[0][0]);
    acc[0][1] = wmma_f16(a0, b1, acc[0][1]);
    acc[0][2] = wmma_f16(a0, b2, acc[0][2]);
    acc[0][3] = wmma_f16(a0, b3, acc[0][3]);
    acc[1][0] = wmma_f16(a1, b0, acc[1][0]);
    acc[1][1] = wmma_f16(a1, b1, acc[1][1]);
    acc[1][2] = wmma_f16(a1, b2, acc[1][2]);
    acc[1][3] = wmma_f16(a1, b3, acc[1][3]);
    wait_async0();       // next-tile copies complete
    __syncthreads();     // all waves done reading buf / done copying
    buf = nxt;
  }

  // ------------------------------ epilogue ------------------------------
  const int nloc  = lane & 15;
  const int rbase = (lane >> 4) << 3;      // C layout: m = rbase + r, n = nloc
  const int mbase = mtile + mw;
  const int nbase = ntile + nw;
#pragma unroll
  for (int i = 0; i < 2; ++i) {
#pragma unroll
    for (int j = 0; j < 4; ++j) {
      const int n  = nbase + 16 * j + nloc;
      const float bv = bias ? bias[n] : 0.0f;
      if (outT) {
        // n -> (h,d); m -> (b,t). 8 consecutive t per lane -> one 16B store.
        const int h  = n >> 6, d = n & 63;
        const int m0 = mbase + 16 * i + rbase;
        const int bb = m0 >> 11, t0 = m0 & 2047;
        v8h pk;
#pragma unroll
        for (int r = 0; r < 8; ++r)
          pk[r] = (_Float16)((acc[i][j][r] + bv) * scale);
        *(v8h*)(outT + (((size_t)(bb * HEADS + h) * HD + d) << 11) + t0) = pk;
      } else {
#pragma unroll
        for (int r = 0; r < 8; ++r) {
          const int m = mbase + 16 * i + rbase + r;
          float v = (acc[i][j][r] + bv) * scale;
          if (do_gelu) v = gelu_exact(v);
          const size_t idx = (size_t)m * ldo + n;
          if (out32) {
            float o = v;
            if (res0) o += res0[idx];
            if (res1) o += res1[idx];
            out32[idx] = o;
          }
          if (out16) out16[idx] = (_Float16)v;
        }
      }
    }
  }
}

// ------------- stage 1: P[bh][dv][dk] = sum_t V[t,dv] * K[t,dk] -------------
// Vt/Kt are [bh][64][2048] f16 (t-contiguous). One block (4 waves) per bh.
__global__ __launch_bounds__(128) void attn_kv_kernel(
    const _Float16* __restrict__ Vt, const _Float16* __restrict__ Kt,
    _Float16* __restrict__ Ph) {
  const int bh   = blockIdx.x;
  const int lane = threadIdx.x & 31;
  const int wave = threadIdx.x >> 5;
  const int mbase = (wave >> 1) * 32;   // dv
  const int nbase = (wave & 1) * 32;    // dk
  const _Float16* Abh = Vt + (size_t)bh * HD * SEQ;
  const _Float16* Wbh = Kt + (size_t)bh * HD * SEQ;

  v8f acc[2][2] = {};
  for (int k0 = 0; k0 < SEQ; k0 += 32) {
    v16h a0 = a_frag_ld(Abh + k0, SEQ, mbase,      lane);
    v16h a1 = a_frag_ld(Abh + k0, SEQ, mbase + 16, lane);
    v16h b0 = b_frag_ld(Wbh + k0, SEQ, nbase,      lane);
    v16h b1 = b_frag_ld(Wbh + k0, SEQ, nbase + 16, lane);
    acc[0][0] = wmma_f16(a0, b0, acc[0][0]);
    acc[0][1] = wmma_f16(a0, b1, acc[0][1]);
    acc[1][0] = wmma_f16(a1, b0, acc[1][0]);
    acc[1][1] = wmma_f16(a1, b1, acc[1][1]);
  }
  const int nloc  = lane & 15;
  const int rbase = (lane >> 4) << 3;
  _Float16* Pbh = Ph + (size_t)bh * HD * HD;
#pragma unroll
  for (int i = 0; i < 2; ++i)
#pragma unroll
    for (int j = 0; j < 2; ++j)
#pragma unroll
      for (int r = 0; r < 8; ++r) {
        const int m = mbase + 16 * i + rbase + r;
        const int n = nbase + 16 * j + nloc;
        Pbh[m * HD + n] = (_Float16)acc[i][j][r];
      }
}

// --------- stage 2: out[t, h*64+dv] = sum_dk Q[t, h*64+dk] * P[dv,dk] -------
__global__ __launch_bounds__(256) void attn_qm_kernel(
    const _Float16* __restrict__ Qh, const _Float16* __restrict__ Ph,
    float* __restrict__ attn32, _Float16* __restrict__ attn16) {
  const int bh   = blockIdx.y;           // 0..31
  const int b    = bh >> 4, h = bh & 15;
  const int lane = threadIdx.x & 31;
  const int wave = threadIdx.x >> 5;
  const int mloc = blockIdx.x * 256 + wave * 32;     // token within batch b

  const _Float16* Ab = Qh + ((size_t)b * SEQ) * DIMC + h * HD;  // lda = DIMC
  const _Float16* Wb = Ph + (size_t)bh * HD * HD;               // ldw = HD

  v8f acc[2][4] = {};
  for (int k0 = 0; k0 < HD; k0 += 32) {
    v16h a0 = a_frag_ld(Ab + k0, DIMC, mloc,      lane);
    v16h a1 = a_frag_ld(Ab + k0, DIMC, mloc + 16, lane);
    v16h b0 = b_frag_ld(Wb + k0, HD, 0,  lane);
    v16h b1 = b_frag_ld(Wb + k0, HD, 16, lane);
    v16h b2 = b_frag_ld(Wb + k0, HD, 32, lane);
    v16h b3 = b_frag_ld(Wb + k0, HD, 48, lane);
    acc[0][0] = wmma_f16(a0, b0, acc[0][0]);
    acc[0][1] = wmma_f16(a0, b1, acc[0][1]);
    acc[0][2] = wmma_f16(a0, b2, acc[0][2]);
    acc[0][3] = wmma_f16(a0, b3, acc[0][3]);
    acc[1][0] = wmma_f16(a1, b0, acc[1][0]);
    acc[1][1] = wmma_f16(a1, b1, acc[1][1]);
    acc[1][2] = wmma_f16(a1, b2, acc[1][2]);
    acc[1][3] = wmma_f16(a1, b3, acc[1][3]);
  }
  const int nloc  = lane & 15;
  const int rbase = (lane >> 4) << 3;
#pragma unroll
  for (int i = 0; i < 2; ++i)
#pragma unroll
    for (int j = 0; j < 4; ++j)
#pragma unroll
      for (int r = 0; r < 8; ++r) {
        const int m = mloc + 16 * i + rbase + r;
        const int n = 16 * j + nloc;
        const float v = acc[i][j][r];
        const size_t idx = ((size_t)b * SEQ + m) * DIMC + h * HD + n;
        attn32[idx] = v;
        attn16[idx] = (_Float16)v;
      }
}

// ---------------------------------------------------------------------------
extern "C" void kernel_launch(void* const* d_in, const int* in_sizes, int n_in,
                              void* d_out, int out_size, void* d_ws, size_t ws_size,
                              hipStream_t stream) {
  const float* x  = (const float*)d_in[0];
  const float* Wq = (const float*)d_in[1];
  const float* bq = (const float*)d_in[2];
  const float* Wk = (const float*)d_in[3];
  const float* bk = (const float*)d_in[4];
  const float* Wv = (const float*)d_in[5];
  const float* bv = (const float*)d_in[6];
  const float* W1 = (const float*)d_in[7];
  const float* b1 = (const float*)d_in[8];
  const float* W2 = (const float*)d_in[9];
  const float* b2 = (const float*)d_in[10];
  float* y = (float*)d_out;

  // -------- workspace carve (all 256B aligned), total ~74.5 MB --------
  char* p = (char*)d_ws;
  auto take = [&](size_t bytes) -> void* {
    void* r = (void*)p;
    p += (bytes + 255) & ~(size_t)255;
    return r;
  };
  _Float16* xh   = (_Float16*)take((size_t)TOK * DIMC * 2);
  _Float16* Wqh  = (_Float16*)take((size_t)DIMC * DIMC * 2);
  _Float16* Wkh  = (_Float16*)take((size_t)DIMC * DIMC * 2);
  _Float16* Wvh  = (_Float16*)take((size_t)DIMC * DIMC * 2);
  _Float16* W1h  = (_Float16*)take((size_t)DIMC * DIMC * 2);
  _Float16* W2h  = (_Float16*)take((size_t)DIMC * DIMC * 2);
  _Float16* Qf   = (_Float16*)take((size_t)TOK * DIMC * 2);
  _Float16* Kt   = (_Float16*)take((size_t)TOK * DIMC * 2);   // [b][h][d][t]
  _Float16* Vt   = (_Float16*)take((size_t)TOK * DIMC * 2);   // [b][h][d][t]
  _Float16* Ph   = (_Float16*)take((size_t)BATCH * HEADS * HD * HD * 2);
  _Float16* At16 = (_Float16*)take((size_t)TOK * DIMC * 2);   // attn out f16
  float*    At32 = (float*)   take((size_t)TOK * DIMC * 4);   // attn out f32
  _Float16* Hh   = (_Float16*)take((size_t)TOK * DIMC * 2);   // gelu(mlp1) f16

  // -------- 1) precision conversion --------
  cvt_f32_to_f16<<<1024, 256, 0, stream>>>(x,  xh,  TOK * DIMC);
  cvt_f32_to_f16<<<512,  256, 0, stream>>>(Wq, Wqh, DIMC * DIMC);
  cvt_f32_to_f16<<<512,  256, 0, stream>>>(Wk, Wkh, DIMC * DIMC);
  cvt_f32_to_f16<<<512,  256, 0, stream>>>(Wv, Wvh, DIMC * DIMC);
  cvt_f32_to_f16<<<512,  256, 0, stream>>>(W1, W1h, DIMC * DIMC);
  cvt_f32_to_f16<<<512,  256, 0, stream>>>(W2, W2h, DIMC * DIMC);

  const dim3 gGemm(DIMC / 128, TOK / 128);  // (8, 32)

  // -------- 2) projections --------
  gemm_wmma_128x128<<<gGemm, 256, 0, stream>>>(           // Q (scaled), f16
      xh, DIMC, Wqh, DIMC, bq, QSCALE, DIMC, DIMC, 0,
      Qf, nullptr, nullptr, nullptr, nullptr);
  gemm_wmma_128x128<<<gGemm, 256, 0, stream>>>(           // K -> [b][h][d][t]
      xh, DIMC, Wkh, DIMC, bk, 1.0f, DIMC, DIMC, 0,
      nullptr, nullptr, nullptr, nullptr, Kt);
  gemm_wmma_128x128<<<gGemm, 256, 0, stream>>>(           // V -> [b][h][d][t]
      xh, DIMC, Wvh, DIMC, bv, 1.0f, DIMC, DIMC, 0,
      nullptr, nullptr, nullptr, nullptr, Vt);

  // -------- 3) attention via associativity: P = V^T K; out = Q P^T --------
  attn_kv_kernel<<<BATCH * HEADS, 128, 0, stream>>>(Vt, Kt, Ph);
  attn_qm_kernel<<<dim3(SEQ / 256, BATCH * HEADS), 256, 0, stream>>>(
      Qf, Ph, At32, At16);

  // -------- 4) MLP --------
  gemm_wmma_128x128<<<gGemm, 256, 0, stream>>>(           // gelu(mlp1) -> f16
      At16, DIMC, W1h, DIMC, b1, 1.0f, DIMC, DIMC, 1,
      Hh, nullptr, nullptr, nullptr, nullptr);
  gemm_wmma_128x128<<<gGemm, 256, 0, stream>>>(           // y = x+out+mlp2
      Hh, DIMC, W2h, DIMC, b2, 1.0f, DIMC, DIMC, 0,
      nullptr, y, x, At32, nullptr);
}
